// Prdnet_3324304687823
// MI455X (gfx1250) — compile-verified, matrix operand order checked by reading, not run on hardware
//
#include <hip/hip_runtime.h>

#define NN 8192      // nodes
#define NE 131072    // edges
#define NB 256       // graphs
#define DD 256       // feature dim
#define NH 8         // heads
#define HDIM 32      // head dim
#define NL 6         // layers
#define FCD 512
#define NBINS 256

typedef __attribute__((ext_vector_type(16))) __bf16 v16bf;
typedef __attribute__((ext_vector_type(8)))  float  v8f;
typedef __attribute__((ext_vector_type(4)))  unsigned int u32x4;

union FragBF { v16bf v; u32x4 q[2]; };

__device__ __forceinline__ unsigned short f2bf(float f) {
  unsigned int u = __float_as_uint(f);
  unsigned int r = ((u >> 16) & 1u) + 0x7FFFu;   // round-to-nearest-even
  return (unsigned short)((u + r) >> 16);
}

// ---------------------------------------------------------------------------
// Weight transpose + fp32 -> bf16:  w[K,N] -> wt[N,K]
// ---------------------------------------------------------------------------
__global__ void wtrans_kernel(const float* __restrict__ w,
                              unsigned short* __restrict__ wt, int K, int N) {
  int idx = blockIdx.x * blockDim.x + threadIdx.x;
  if (idx >= K * N) return;
  int k = idx / N, n = idx % N;
  wt[(size_t)n * K + k] = f2bf(w[idx]);
}

// ---------------------------------------------------------------------------
// WMMA GEMM (K fixed = 256):
//   C[M,Nout](f32) = A[M,256](bf16) * BT[Nout,256](bf16)^T + bias
// block = 256 threads = 8 waves; block tile 128x64.
// B panel (64x256 bf16 = 32KB) staged in LDS via async-to-LDS copies and
// shared by all 8 waves; each wave owns a 16-row strip and 4 N-tiles
// (16x64 of C -> 4 v_wmma per 32-wide K step, single A fragment reused).
// All four B fragments are loaded before the WMMA quartet so the ds_loads
// batch up and the waits are partial instead of dscnt==0 per WMMA.
// ---------------------------------------------------------------------------
__global__ __launch_bounds__(256)
void gemm_wmma_bf16(const unsigned short* __restrict__ A,
                    const unsigned short* __restrict__ BT,
                    const float* __restrict__ bias,
                    float* __restrict__ C, int M, int Nout) {
  __shared__ unsigned short sB[64 * 256];          // 32 KB B panel

  const int lane  = threadIdx.x & 31;
  const int wave  = threadIdx.x >> 5;
  const int half  = lane >> 4;                     // 16-lane half
  const int l15   = lane & 15;
  const int blockM = blockIdx.x * 128;
  const int blockN = blockIdx.y * 64;

  // ---- async-stage the 64x256 bf16 B panel (contiguous in BT) into LDS ----
  // NOTE: the LDS address operand is derived from &sB[...] so that sB escapes
  // into the asm (memory clobber) -- otherwise LLVM proves the asm cannot
  // write sB and folds the later ds_loads of "uninitialized" LDS to undef.
  {
    const unsigned short* gB = BT + (size_t)blockN * 256;
#pragma unroll
    for (int i = 0; i < 8; ++i) {
      int chunk = threadIdx.x + i * 256;           // 0..2047 x 16B = 32KB
      unsigned lds_off = (unsigned)(size_t)(&sB[chunk * 8]);  // wave-rel LDS byte addr
      unsigned long long ga =
          (unsigned long long)(const void*)(gB + chunk * 8);
      asm volatile("global_load_async_to_lds_b128 %0, %1, off"
                   :: "v"(lds_off), "v"(ga) : "memory");
    }
    asm volatile("s_wait_asynccnt 0x0" ::: "memory");
    __syncthreads();
  }

  const int row = blockM + wave * 16 + l15;
  const unsigned short* arow = A + (size_t)row * 256;

  v8f c[4];
#pragma unroll
  for (int nt = 0; nt < 4; ++nt)
    c[nt] = (v8f){0.f, 0.f, 0.f, 0.f, 0.f, 0.f, 0.f, 0.f};

  for (int kt = 0; kt < 256; kt += 32) {
    const int base = kt + half * 8;
    FragBF a;
    a.q[0] = *(const u32x4*)(arow + base);
    a.q[1] = *(const u32x4*)(arow + base + 16);
    FragBF b[4];
#pragma unroll
    for (int nt = 0; nt < 4; ++nt) {
      const unsigned short* bp = sB + (size_t)(nt * 16 + l15) * 256 + base;
      b[nt].q[0] = *(const u32x4*)(bp);
      b[nt].q[1] = *(const u32x4*)(bp + 16);
    }
#pragma unroll
    for (int nt = 0; nt < 4; ++nt) {
      c[nt] = __builtin_amdgcn_wmma_f32_16x16x32_bf16(false, a.v, false, b[nt].v,
                                                      (short)0, c[nt], false, false);
    }
  }

  const int srow0 = blockM + wave * 16 + half * 8;   // C VGPR r -> row srow0+r
#pragma unroll
  for (int nt = 0; nt < 4; ++nt) {
    const int col = blockN + nt * 16 + l15;
    const float bc = bias ? bias[col] : 0.f;
#pragma unroll
    for (int r = 0; r < 8; ++r)
      C[(size_t)(srow0 + r) * Nout + col] = c[nt][r] + bc;
  }
}

// ---------------------------------------------------------------------------
// Row post-process: optional LayerNorm(g,b), optional SiLU, optional residual,
// writes optional f32 and optional bf16 copies. One wave32 per 256-wide row.
// ---------------------------------------------------------------------------
__global__ __launch_bounds__(256)
void postproc_kernel(const float* __restrict__ src, const float* __restrict__ res,
                     const float* __restrict__ gamma, const float* __restrict__ beta,
                     int do_silu, float* __restrict__ out_f32,
                     unsigned short* __restrict__ out_bf, int rows) {
  int gw   = (blockIdx.x * blockDim.x + threadIdx.x) >> 5;
  int lane = threadIdx.x & 31;
  if (gw >= rows) return;
  const float* x = src + (size_t)gw * DD;
  float v[8], sum = 0.f, sumsq = 0.f;
#pragma unroll
  for (int i = 0; i < 8; ++i) {
    v[i] = x[lane + i * 32];
    sum += v[i]; sumsq += v[i] * v[i];
  }
  if (gamma) {
    for (int off = 16; off >= 1; off >>= 1) {
      sum   += __shfl_xor(sum, off, 32);
      sumsq += __shfl_xor(sumsq, off, 32);
    }
    float mu  = sum * (1.f / DD);
    float var = sumsq * (1.f / DD) - mu * mu;
    float rstd = rsqrtf(var + 1e-5f);
#pragma unroll
    for (int i = 0; i < 8; ++i) {
      int d = lane + i * 32;
      v[i] = (v[i] - mu) * rstd * gamma[d] + beta[d];
    }
  }
  if (do_silu) {
#pragma unroll
    for (int i = 0; i < 8; ++i) v[i] = v[i] / (1.f + __expf(-v[i]));
  }
  if (res) {
    const float* r = res + (size_t)gw * DD;
#pragma unroll
    for (int i = 0; i < 8; ++i) v[i] += r[lane + i * 32];
  }
  if (out_f32) {
    float* o = out_f32 + (size_t)gw * DD;
#pragma unroll
    for (int i = 0; i < 8; ++i) o[lane + i * 32] = v[i];
  }
  if (out_bf) {
    unsigned short* o = out_bf + (size_t)gw * DD;
#pragma unroll
    for (int i = 0; i < 8; ++i) o[lane + i * 32] = f2bf(v[i]);
  }
}

// ---------------------------------------------------------------------------
// Atom embedding first layer: h = x[NN,92] @ aw1[92,256] + ab1  (K=92, scalar)
// ---------------------------------------------------------------------------
__global__ void atom_fc1_kernel(const float* __restrict__ x,
                                const float* __restrict__ w,
                                const float* __restrict__ b,
                                float* __restrict__ out) {
  int idx = blockIdx.x * blockDim.x + threadIdx.x;
  if (idx >= NN * DD) return;
  int n = idx >> 8, d = idx & 255;
  float acc = b[d];
  const float* xr = x + n * 92;
#pragma unroll 4
  for (int k = 0; k < 92; ++k) acc += xr[k] * w[k * DD + d];
  out[idx] = acc;
}

// ---------------------------------------------------------------------------
// RBF expansion of edge lengths -> bf16 [NE, NBINS]
// ---------------------------------------------------------------------------
__global__ void rbf_kernel(const float* __restrict__ edge_attr,
                           unsigned short* __restrict__ out_bf) {
  int idx = blockIdx.x * blockDim.x + threadIdx.x;
  if (idx >= NE * NBINS) return;
  int e = idx >> 8, bin = idx & 255;
  float dx = edge_attr[e * 3 + 0], dy = edge_attr[e * 3 + 1], dz = edge_attr[e * 3 + 2];
  float d = sqrtf(dx * dx + dy * dy + dz * dz);
  const float step  = 8.0f / 255.0f;
  const float gamma = 1.0f / (step * step);
  float t = d - step * (float)bin;
  out_bf[idx] = f2bf(__expf(-gamma * t * t));
}

__global__ void fillf_kernel(float* __restrict__ p, float v, int n) {
  int i = blockIdx.x * blockDim.x + threadIdx.x;
  if (i < n) p[i] = v;
}

__device__ __forceinline__ void atomicMaxF(float* addr, float v) {
  if (v >= 0.f) atomicMax((int*)addr, __float_as_int(v));
  else          atomicMin((unsigned int*)addr, __float_as_uint(v));
}

// score[e,h] = sum_d q[dst,h,d]*(k[src,h,d]+ee[e,h,d]) * inv_sqrt; smax = seg max
__global__ void score_kernel(const float* __restrict__ q, const float* __restrict__ k,
                             const float* __restrict__ ee, const int* __restrict__ src,
                             const int* __restrict__ dst, float* __restrict__ score,
                             float* __restrict__ smax) {
  int idx = blockIdx.x * blockDim.x + threadIdx.x;
  if (idx >= NE * NH) return;
  int e = idx >> 3, h = idx & 7;
  int s = src[e], d = dst[e];
  const float* qp = q  + (size_t)d * DD + h * HDIM;
  const float* kp = k  + (size_t)s * DD + h * HDIM;
  const float* ep = ee + (size_t)e * DD + h * HDIM;
  float acc = 0.f;
#pragma unroll
  for (int i = 0; i < HDIM; ++i) acc += qp[i] * (kp[i] + ep[i]);
  acc *= 0.17677669529663687f;   // 1/sqrt(32)
  score[idx] = acc;
  atomicMaxF(&smax[d * NH + h], acc);
}

__global__ void expsum_kernel(float* __restrict__ score, const float* __restrict__ smax,
                              float* __restrict__ den, const int* __restrict__ dst) {
  int idx = blockIdx.x * blockDim.x + threadIdx.x;
  if (idx >= NE * NH) return;
  int e = idx >> 3, h = idx & 7;
  int d = dst[e];
  float ex = __expf(score[idx] - smax[d * NH + h]);
  score[idx] = ex;
  atomicAdd(&den[d * NH + h], ex);
}

__global__ void message_kernel(const float* __restrict__ score, const float* __restrict__ den,
                               const float* __restrict__ v, const float* __restrict__ ee,
                               const int* __restrict__ src, const int* __restrict__ dst,
                               float* __restrict__ msg) {
  long long idx = (long long)blockIdx.x * blockDim.x + threadIdx.x;
  if (idx >= (long long)NE * DD) return;
  int e = (int)(idx >> 8), dd = (int)(idx & 255);
  int h = dd >> 5;
  int s = src[e], d = dst[e];
  float alpha = score[e * NH + h] / (den[d * NH + h] + 1e-16f);
  float vj = v[(size_t)s * DD + dd] + ee[(size_t)e * DD + dd];
  atomicAdd(&msg[(size_t)d * DD + dd], alpha * vj);
}

__global__ void pool_sum_kernel(const float* __restrict__ node, const int* __restrict__ batch,
                                float* __restrict__ sums) {
  int idx = blockIdx.x * blockDim.x + threadIdx.x;
  if (idx >= NN * DD) return;
  int n = idx >> 8, d = idx & 255;
  atomicAdd(&sums[(size_t)batch[n] * DD + d], node[idx]);
}

__global__ void pool_cnt_kernel(const int* __restrict__ batch, float* __restrict__ cnt) {
  int n = blockIdx.x * blockDim.x + threadIdx.x;
  if (n < NN) atomicAdd(&cnt[batch[n]], 1.0f);
}

__global__ void feats_kernel(const float* __restrict__ sums, const float* __restrict__ cnt,
                             unsigned short* __restrict__ out_bf) {
  int idx = blockIdx.x * blockDim.x + threadIdx.x;
  if (idx >= NB * DD) return;
  int b = idx >> 8;
  out_bf[idx] = f2bf(sums[idx] / fmaxf(cnt[b], 1.0f));
}

__global__ void fcout_kernel(const float* __restrict__ fc1, const float* __restrict__ fow,
                             const float* __restrict__ fob, float* __restrict__ out) {
  int b = blockIdx.x, lane = threadIdx.x;
  float acc = 0.f;
  for (int j = lane; j < FCD; j += 32) acc += fc1[(size_t)b * FCD + j] * fow[j];
  for (int off = 16; off >= 1; off >>= 1) acc += __shfl_xor(acc, off, 32);
  if (lane == 0) out[b] = acc + fob[0];
}

// ---------------------------------------------------------------------------
extern "C" void kernel_launch(void* const* d_in, const int* in_sizes, int n_in,
                              void* d_out, int out_size, void* d_ws, size_t ws_size,
                              hipStream_t stream) {
  const float* x         = (const float*)d_in[0];
  const float* edge_attr = (const float*)d_in[1];
  const int*   eidx      = (const int*)  d_in[2];
  const int*   batch     = (const int*)  d_in[3];
  const float* aw1 = (const float*)d_in[4],  *ab1 = (const float*)d_in[5];
  const float* ag1 = (const float*)d_in[6],  *abt1= (const float*)d_in[7];
  const float* aw2 = (const float*)d_in[8],  *ab2 = (const float*)d_in[9];
  const float* rw1 = (const float*)d_in[10], *rb1 = (const float*)d_in[11];
  const float* rg1 = (const float*)d_in[12], *rbt1= (const float*)d_in[13];
  const float* rw2 = (const float*)d_in[14], *rb2 = (const float*)d_in[15];
  const float* rg2 = (const float*)d_in[16], *rbt2= (const float*)d_in[17];
  const float* rw3 = (const float*)d_in[18], *rb3 = (const float*)d_in[19];
  const float* Wq  = (const float*)d_in[20], *bq  = (const float*)d_in[21];
  const float* Wk  = (const float*)d_in[22], *bk  = (const float*)d_in[23];
  const float* Wv  = (const float*)d_in[24], *bv  = (const float*)d_in[25];
  const float* We  = (const float*)d_in[26], *be  = (const float*)d_in[27];
  const float* Wo  = (const float*)d_in[28], *bo  = (const float*)d_in[29];
  const float* lng = (const float*)d_in[30], *lnb = (const float*)d_in[31];
  const float* fw1 = (const float*)d_in[32], *fb1 = (const float*)d_in[33];
  const float* fow = (const float*)d_in[34], *fob = (const float*)d_in[35];

  const int src_off = 0, dst_off = NE;

  // ---- workspace carve-up ----
  char* ws = (char*)d_ws;
  size_t off = 0;
  auto alloc = [&](size_t bytes) -> char* {
    char* p = ws + off;
    off += (bytes + 255) & ~(size_t)255;
    return p;
  };
  unsigned short* aw2T = (unsigned short*)alloc(DD * DD * 2);
  unsigned short* rw1T = (unsigned short*)alloc(DD * DD * 2);
  unsigned short* rw2T = (unsigned short*)alloc(DD * DD * 2);
  unsigned short* rw3T = (unsigned short*)alloc(DD * DD * 2);
  unsigned short* WqT  = (unsigned short*)alloc((size_t)NL * DD * DD * 2);
  unsigned short* WkT  = (unsigned short*)alloc((size_t)NL * DD * DD * 2);
  unsigned short* WvT  = (unsigned short*)alloc((size_t)NL * DD * DD * 2);
  unsigned short* WeT  = (unsigned short*)alloc((size_t)NL * DD * DD * 2);
  unsigned short* WoT  = (unsigned short*)alloc((size_t)NL * DD * DD * 2);
  unsigned short* fw1T = (unsigned short*)alloc((size_t)FCD * DD * 2);

  float* nodeA   = (float*)alloc((size_t)NN * DD * 4);
  float* nodeB   = (float*)alloc((size_t)NN * DD * 4);
  float* tmpn    = (float*)alloc((size_t)NN * DD * 4);
  unsigned short* node_bf = (unsigned short*)alloc((size_t)NN * DD * 2);
  unsigned short* msg_bf  = (unsigned short*)alloc((size_t)NN * DD * 2);
  float* qf = (float*)alloc((size_t)NN * DD * 4);
  float* kf = (float*)alloc((size_t)NN * DD * 4);
  float* vf = (float*)alloc((size_t)NN * DD * 4);
  float* msg = (float*)alloc((size_t)NN * DD * 4);

  float* etmp = (float*)alloc((size_t)NE * DD * 4);           // 128 MB, reused as ee
  unsigned short* ebfA = (unsigned short*)alloc((size_t)NE * DD * 2);
  unsigned short* ebfB = (unsigned short*)alloc((size_t)NE * DD * 2);

  float* score = (float*)alloc((size_t)NE * NH * 4);
  float* smax  = (float*)alloc((size_t)NN * NH * 4);
  float* den   = (float*)alloc((size_t)NN * NH * 4);
  float* sums  = (float*)alloc((size_t)NB * DD * 4);
  float* cnt   = (float*)alloc((size_t)NB * 4);
  float* fc1f  = (float*)alloc((size_t)NB * FCD * 4);
  unsigned short* feats_bf = (unsigned short*)alloc((size_t)NB * DD * 2);

  const int TB = 256;
  const int WG = DD * DD / TB;     // 256 blocks for a 256x256 transpose

  // ---- weight prep (bf16, transposed to [N,K]) ----
  wtrans_kernel<<<WG, TB, 0, stream>>>(aw2, aw2T, DD, DD);
  wtrans_kernel<<<WG, TB, 0, stream>>>(rw1, rw1T, DD, DD);
  wtrans_kernel<<<WG, TB, 0, stream>>>(rw2, rw2T, DD, DD);
  wtrans_kernel<<<WG, TB, 0, stream>>>(rw3, rw3T, DD, DD);
  for (int l = 0; l < NL; ++l) {
    size_t o = (size_t)l * DD * DD;
    wtrans_kernel<<<WG, TB, 0, stream>>>(Wq + o, WqT + o, DD, DD);
    wtrans_kernel<<<WG, TB, 0, stream>>>(Wk + o, WkT + o, DD, DD);
    wtrans_kernel<<<WG, TB, 0, stream>>>(Wv + o, WvT + o, DD, DD);
    wtrans_kernel<<<WG, TB, 0, stream>>>(We + o, WeT + o, DD, DD);
    wtrans_kernel<<<WG, TB, 0, stream>>>(Wo + o, WoT + o, DD, DD);
  }
  wtrans_kernel<<<DD * FCD / TB, TB, 0, stream>>>(fw1, fw1T, DD, FCD);

  dim3 gN(NN / 128, DD / 64);      // node GEMM grid
  dim3 gE(NE / 128, DD / 64);      // edge GEMM grid

  // ---- atom embedding ----
  atom_fc1_kernel<<<NN * DD / TB, TB, 0, stream>>>(x, aw1, ab1, tmpn);
  postproc_kernel<<<NN / 8, TB, 0, stream>>>(tmpn, nullptr, ag1, abt1, 1, nullptr, node_bf, NN);
  gemm_wmma_bf16<<<gN, TB, 0, stream>>>(node_bf, aw2T, ab2, nodeA, NN, DD);
  postproc_kernel<<<NN / 8, TB, 0, stream>>>(nodeA, nullptr, nullptr, nullptr, 0, nullptr, node_bf, NN);

  // ---- RBF + edge MLP ----
  rbf_kernel<<<NE * NBINS / TB, TB, 0, stream>>>(edge_attr, ebfA);
  gemm_wmma_bf16<<<gE, TB, 0, stream>>>(ebfA, rw1T, rb1, etmp, NE, DD);
  postproc_kernel<<<NE / 8, TB, 0, stream>>>(etmp, nullptr, rg1, rbt1, 1, nullptr, ebfB, NE);
  gemm_wmma_bf16<<<gE, TB, 0, stream>>>(ebfB, rw2T, rb2, etmp, NE, DD);
  postproc_kernel<<<NE / 8, TB, 0, stream>>>(etmp, nullptr, rg2, rbt2, 1, nullptr, ebfA, NE);
  gemm_wmma_bf16<<<gE, TB, 0, stream>>>(ebfA, rw3T, rb3, etmp, NE, DD);
  postproc_kernel<<<NE / 8, TB, 0, stream>>>(etmp, nullptr, nullptr, nullptr, 0, nullptr, ebfB, NE);
  // ebfB now holds e (bf16) for all layers

  float* nodeCur = nodeA;
  float* nodeNxt = nodeB;

  for (int l = 0; l < NL; ++l) {
    size_t wo = (size_t)l * DD * DD;
    size_t bo_ = (size_t)l * DD;
    gemm_wmma_bf16<<<gN, TB, 0, stream>>>(node_bf, WqT + wo, bq + bo_, qf, NN, DD);
    gemm_wmma_bf16<<<gN, TB, 0, stream>>>(node_bf, WkT + wo, bk + bo_, kf, NN, DD);
    gemm_wmma_bf16<<<gN, TB, 0, stream>>>(node_bf, WvT + wo, bv + bo_, vf, NN, DD);
    gemm_wmma_bf16<<<gE, TB, 0, stream>>>(ebfB, WeT + wo, be + bo_, etmp, NE, DD);

    fillf_kernel<<<(NN * NH + TB - 1) / TB, TB, 0, stream>>>(smax, -3.0e38f, NN * NH);
    fillf_kernel<<<(NN * NH + TB - 1) / TB, TB, 0, stream>>>(den, 0.f, NN * NH);
    fillf_kernel<<<(NN * DD + TB - 1) / TB, TB, 0, stream>>>(msg, 0.f, NN * DD);

    score_kernel<<<NE * NH / TB, TB, 0, stream>>>(qf, kf, etmp, eidx + src_off, eidx + dst_off, score, smax);
    expsum_kernel<<<NE * NH / TB, TB, 0, stream>>>(score, smax, den, eidx + dst_off);
    message_kernel<<<NE * DD / TB, TB, 0, stream>>>(score, den, vf, etmp, eidx + src_off, eidx + dst_off, msg);

    postproc_kernel<<<NN / 8, TB, 0, stream>>>(msg, nullptr, nullptr, nullptr, 0, nullptr, msg_bf, NN);
    gemm_wmma_bf16<<<gN, TB, 0, stream>>>(msg_bf, WoT + wo, bo + bo_, tmpn, NN, DD);
    // out = LN(out); node = out + res   (f32 + bf16 copies)
    postproc_kernel<<<NN / 8, TB, 0, stream>>>(tmpn, nodeCur, lng + bo_, lnb + bo_, 0, nodeNxt, node_bf, NN);
    float* t = nodeCur; nodeCur = nodeNxt; nodeNxt = t;
  }

  // ---- scatter-mean pooling + FC head ----
  fillf_kernel<<<(NB * DD + TB - 1) / TB, TB, 0, stream>>>(sums, 0.f, NB * DD);
  fillf_kernel<<<(NB + TB - 1) / TB, TB, 0, stream>>>(cnt, 0.f, NB);
  pool_sum_kernel<<<NN * DD / TB, TB, 0, stream>>>(nodeCur, batch, sums);
  pool_cnt_kernel<<<NN / TB, TB, 0, stream>>>(batch, cnt);
  feats_kernel<<<NB * DD / TB, TB, 0, stream>>>(sums, cnt, feats_bf);
  gemm_wmma_bf16<<<dim3(NB / 128, FCD / 64), TB, 0, stream>>>(feats_bf, fw1T, fb1, fc1f, NB, DD);
  fcout_kernel<<<NB, 32, 0, stream>>>(fc1f, fow, fob, (float*)d_out);
}
